// CrossAttention_44753559224708
// MI455X (gfx1250) — matrix operand order repaired and checked by reading, not compiled, not run on hardware
//
#include <hip/hip_runtime.h>

// ---------------------------------------------------------------------------
// CDNA5 / gfx1250 cross-attention: bf16 WMMA + double-buffered TDM staging.
// ---------------------------------------------------------------------------

typedef __attribute__((ext_vector_type(16))) __bf16 v16bf;
typedef __attribute__((ext_vector_type(8)))  __bf16 v8bf;
typedef __attribute__((ext_vector_type(8)))  float  v8f;
typedef __attribute__((ext_vector_type(4)))  unsigned int v4u;
typedef __attribute__((ext_vector_type(8)))  int v8i;
typedef __attribute__((ext_vector_type(4)))  int v4i;

#define WMMA_BF16(A_, B_, C_) \
  __builtin_amdgcn_wmma_f32_16x16x32_bf16(false, (A_), false, (B_), (short)0, (C_), false, false)

// Problem constants
static constexpr int BATCH = 2;
static constexpr int TQ    = 2048;
static constexpr int TKV   = 2048;
static constexpr int CEMB  = 1024;
static constexpr int NHEAD = 16;
static constexpr int DHEAD = 64;

// ---------------------------------------------------------------------------
// DPP16 row_ror butterfly all-reduce over each 16-lane group (wave32 rows).
// ---------------------------------------------------------------------------
template <int CTRL>
__device__ __forceinline__ float dpp_rorf(float x) {
  return __builtin_bit_cast(
      float, __builtin_amdgcn_update_dpp(0, __builtin_bit_cast(int, x),
                                         CTRL, 0xF, 0xF, true));
}
__device__ __forceinline__ float reduce16_max(float v) {
  v = fmaxf(v, dpp_rorf<0x121>(v));   // row_ror:1
  v = fmaxf(v, dpp_rorf<0x122>(v));   // row_ror:2
  v = fmaxf(v, dpp_rorf<0x124>(v));   // row_ror:4
  v = fmaxf(v, dpp_rorf<0x128>(v));   // row_ror:8
  return v;
}
__device__ __forceinline__ float reduce16_sum(float v) {
  v += dpp_rorf<0x121>(v);
  v += dpp_rorf<0x122>(v);
  v += dpp_rorf<0x124>(v);
  v += dpp_rorf<0x128>(v);
  return v;
}

// ---------------------------------------------------------------------------
// TDM: async-load a 2D bf16 tile [tile_h x tile_w] (elements) whose rows are
// stride0 elements apart in global memory, into LDS packed rows (tile_w*2 B
// per row). D# packing per CDNA5 ISA §8.3/8.4. EXEC is ignored by the TDM,
// so only one wave per block may execute this. Tracked by TENSORcnt.
// ---------------------------------------------------------------------------
__device__ __forceinline__ void tdm_load_2d(unsigned lds_off, const void* gptr,
                                            int tile_w, int tile_h,
                                            int stride0) {
  unsigned long long ga = (unsigned long long)(uintptr_t)gptr;
  // Group 0: count=1 | lds_addr | global_addr[56:0] | type=2
  v4u g0;
  g0[0] = 1u;
  g0[1] = lds_off;
  g0[2] = (unsigned)ga;
  g0[3] = ((unsigned)((ga >> 32) & 0x01FFFFFFull)) | (2u << 30);
  // Group 1: wg_mask=0, data_size=1 (2B), no pad/iterate;
  //          tensor dims == tile dims (tiles never cross the tensor edge).
  v8i g1;
  g1[0] = 0x00010000;                                       // data_size=2B
  g1[1] = (tile_w & 0xFFFF) << 16;                          // tensor_dim0 lo
  g1[2] = ((tile_w >> 16) & 0xFFFF) | ((tile_h & 0xFFFF) << 16);
  g1[3] = ((tile_h >> 16) & 0xFFFF) | ((tile_w & 0xFFFF) << 16);  // tile_dim0
  g1[4] = tile_h & 0xFFFF;                                  // tile_dim1
  g1[5] = stride0;                                          // dim0_stride lo32
  g1[6] = 0;
  g1[7] = 0;
  v4i gz = {0, 0, 0, 0};
#if __clang_major__ >= 23
  v8i gz8 = {0, 0, 0, 0, 0, 0, 0, 0};
  __builtin_amdgcn_tensor_load_to_lds(g0, g1, gz, gz, gz8, 0);
#else
  __builtin_amdgcn_tensor_load_to_lds(g0, g1, gz, gz, 0);
#endif
}

// ---------------------------------------------------------------------------
// A-fragment loader: 16x32 bf16 tile. Lane half kg: elems 0..15 <-> K=kg*16+e.
// Two 16-byte loads.
// ---------------------------------------------------------------------------
__device__ __forceinline__ v16bf load_a_frag(const __bf16* rowPtr, int kg) {
  v8bf lo = *(const v8bf*)(rowPtr + kg * 16);
  v8bf hi = *(const v8bf*)(rowPtr + kg * 16 + 8);
  v16bf f;
#pragma unroll
  for (int e = 0; e < 8; ++e) { f[e] = lo[e]; f[e + 8] = hi[e]; }
  return f;
}

// ---------------------------------------------------------------------------
// Elementwise f32 -> bf16 convert
// ---------------------------------------------------------------------------
__global__ void cvt_f32_bf16(const float* __restrict__ src,
                             __bf16* __restrict__ dst, int n) {
  int i = blockIdx.x * blockDim.x + threadIdx.x;
  if (i < n) dst[i] = (__bf16)src[i];
}

// W [K x N] row-major  ->  Wt [N x K] row-major, bf16
__global__ void transpose_cvt(const float* __restrict__ W,
                              __bf16* __restrict__ Wt, int K, int N) {
  int i = blockIdx.x * blockDim.x + threadIdx.x;
  if (i < K * N) {
    int k = i / N, n = i % N;
    Wt[(size_t)n * K + k] = (__bf16)W[i];
  }
}

// Build V^T: kvb [B*TKV, 2C] (v part starts at col C) -> vT [B,H,DHEAD,TKV]
__global__ void make_vT(const __bf16* __restrict__ kvb,
                        __bf16* __restrict__ vT) {
  int i = blockIdx.x * blockDim.x + threadIdx.x;
  int total = BATCH * NHEAD * DHEAD * TKV;
  if (i < total) {
    int t = i % TKV;
    int r = i / TKV;
    int d = r % DHEAD; r /= DHEAD;
    int h = r % NHEAD;
    int b = r / NHEAD;
    vT[i] = kvb[((size_t)(b * TKV + t)) * (2 * CEMB) + CEMB + h * DHEAD + d];
  }
}

// ---------------------------------------------------------------------------
// bf16 GEMM, double-buffered TDM-staged LDS tiles:
//   C[MxN] = A[MxK] @ Bt[NxK]^T   (Bt pre-transposed)
// Block = 256 threads = 8 waves; wave -> 32x32 tile (2x2 WMMA accumulators);
// block tile = 128 (M) x 64 (N); K step = 32. Wave 0 drives the TDM; tile
// i+1 streams in while all waves run WMMAs on tile i.
// ---------------------------------------------------------------------------
template <bool OUT_F32>
__global__ __launch_bounds__(256)
void gemm_bf16_wmma(const __bf16* __restrict__ A, const __bf16* __restrict__ Bt,
                    void* __restrict__ Cout, int M, int N, int K) {
  const int lane = threadIdx.x & 31;
  const int wave = threadIdx.x >> 5;
  const int ml = lane & 15;
  const int kg = lane >> 4;
  const int wm = (wave & 3) * 32;   // wave's M offset inside block tile
  const int wn = (wave >> 2) * 32;  // wave's N offset inside block tile
  const int mblk = blockIdx.y * 128;
  const int nblk = blockIdx.x * 64;

  __shared__ __align__(128) __bf16 ldsA[2][128 * 32];
  __shared__ __align__(128) __bf16 ldsB[2][64 * 32];
  unsigned offA[2] = {(unsigned)(uintptr_t)&ldsA[0][0],
                      (unsigned)(uintptr_t)&ldsA[1][0]};
  unsigned offB[2] = {(unsigned)(uintptr_t)&ldsB[0][0],
                      (unsigned)(uintptr_t)&ldsB[1][0]};

  const __bf16* Ab = A + (size_t)mblk * K;
  const __bf16* Bb = Bt + (size_t)nblk * K;

  v8f acc00 = {}, acc01 = {}, acc10 = {}, acc11 = {};

  if (wave == 0) {  // prologue: stage tile 0
    tdm_load_2d(offA[0], Ab, 32, 128, K);
    tdm_load_2d(offB[0], Bb, 32, 64, K);
  }

  int buf = 0;
  for (int k0 = 0; k0 < K; k0 += 32, buf ^= 1) {
    if (wave == 0) __builtin_amdgcn_s_wait_tensorcnt((short)0);
    __syncthreads();  // tile(k0) visible; prior reads of buf^1 are done
    if (wave == 0 && k0 + 32 < K) {  // stream next tile into other buffer
      tdm_load_2d(offA[buf ^ 1], Ab + k0 + 32, 32, 128, K);
      tdm_load_2d(offB[buf ^ 1], Bb + k0 + 32, 32, 64, K);
    }

    const __bf16* la = ldsA[buf];
    const __bf16* lb = ldsB[buf];
    v16bf fa0 = load_a_frag(la + (wm + ml) * 32, kg);
    v16bf fa1 = load_a_frag(la + (wm + 16 + ml) * 32, kg);
    v16bf fb0 = *(const v16bf*)(lb + (wn + ml) * 32 + kg * 16);
    v16bf fb1 = *(const v16bf*)(lb + (wn + 16 + ml) * 32 + kg * 16);
    acc00 = WMMA_BF16(fa0, fb0, acc00);
    acc01 = WMMA_BF16(fa0, fb1, acc01);
    acc10 = WMMA_BF16(fa1, fb0, acc10);
    acc11 = WMMA_BF16(fa1, fb1, acc11);

    __syncthreads();  // reads of tile(k0) done before it is overwritten
  }

  const int m0 = mblk + wm;
  const int n0 = nblk + wn;
#pragma unroll
  for (int r = 0; r < 8; ++r) {
    const int row0 = m0 + r + 8 * kg;
    const int row1 = row0 + 16;
    const int col0 = n0 + ml;
    const int col1 = col0 + 16;
    if (OUT_F32) {
      float* C = (float*)Cout;
      C[(size_t)row0 * N + col0] = acc00[r];
      C[(size_t)row0 * N + col1] = acc01[r];
      C[(size_t)row1 * N + col0] = acc10[r];
      C[(size_t)row1 * N + col1] = acc11[r];
    } else {
      __bf16* C = (__bf16*)Cout;
      C[(size_t)row0 * N + col0] = (__bf16)acc00[r];
      C[(size_t)row0 * N + col1] = (__bf16)acc01[r];
      C[(size_t)row1 * N + col0] = (__bf16)acc10[r];
      C[(size_t)row1 * N + col1] = (__bf16)acc11[r];
    }
  }
}

// ---------------------------------------------------------------------------
// Flash-attention. grid = (TQ/128, NHEAD, BATCH), block = 256 (8 waves).
// All waves in a block share (b,h) and the KV stream, so the shared
// K[32x64] and V^T[64x32] tiles are TDM-staged into LDS (double-buffered)
// instead of each wave re-loading them from global (8x traffic cut).
// Per wave: 16 query rows, online softmax with DPP16 row reductions.
//   scores: S[16x32] = Q[16x64] @ K^T   (4 WMMA)   -> *8 = /SCALE (ref)
//   PV:     O[16x64] += P[16x32] @ V    (4 WMMA)
// ---------------------------------------------------------------------------
__global__ __launch_bounds__(256)
void attn_wmma(const __bf16* __restrict__ qb,   // [B*TQ, C]
               const __bf16* __restrict__ kvb,  // [B*TKV, 2C], k at col h*64
               const __bf16* __restrict__ vT,   // [B,H,DHEAD,TKV]
               __bf16* __restrict__ yb) {       // [B*TQ, C]
  const int lane = threadIdx.x & 31;
  const int wave = threadIdx.x >> 5;
  const int ml = lane & 15;
  const int kg = lane >> 4;
  const int b = blockIdx.z;
  const int h = blockIdx.y;
  const int q0 = blockIdx.x * 128 + wave * 16;

  __shared__ __align__(128) __bf16 Kst[2][32][64];  // KV tile of K rows
  __shared__ __align__(128) __bf16 Vst[2][64][32];  // V^T tile
  __shared__ __bf16 Pst[8][16][34];                 // per-wave P staging
  unsigned offK[2] = {(unsigned)(uintptr_t)&Kst[0][0][0],
                      (unsigned)(uintptr_t)&Kst[1][0][0]};
  unsigned offV[2] = {(unsigned)(uintptr_t)&Vst[0][0][0],
                      (unsigned)(uintptr_t)&Vst[1][0][0]};

  const __bf16* kbase = kvb + (size_t)(b * TKV) * (2 * CEMB) + h * DHEAD;
  const __bf16* vheadT = vT + (((size_t)(b * NHEAD + h)) * DHEAD) * TKV;

  // Q fragments: rows q0..q0+15, Dh split into K-chunks [0,32) and [32,64)
  v16bf qa0, qa1;
  {
    const __bf16* qr = qb + ((size_t)(b * TQ + q0 + ml)) * CEMB + h * DHEAD;
    qa0 = load_a_frag(qr, kg);
    qa1 = load_a_frag(qr + 32, kg);
  }

  v8f o0 = {}, o1 = {}, o2 = {}, o3 = {};
  float mrow[8], lrow[8];
#pragma unroll
  for (int r = 0; r < 8; ++r) { mrow[r] = -1.0e30f; lrow[r] = 0.0f; }

  if (wave == 0) {  // prologue: stage KV tile 0
    tdm_load_2d(offK[0], kbase, DHEAD, 32, 2 * CEMB);
    tdm_load_2d(offV[0], vheadT, 32, DHEAD, TKV);
  }

  int buf = 0;
  for (int kv0 = 0; kv0 < TKV; kv0 += 32, buf ^= 1) {
    if (wave == 0) __builtin_amdgcn_s_wait_tensorcnt((short)0);
    __syncthreads();  // KV tile(kv0) visible to all waves
    if (wave == 0 && kv0 + 32 < TKV) {  // stream next KV tile
      tdm_load_2d(offK[buf ^ 1], kbase + (size_t)(kv0 + 32) * (2 * CEMB),
                  DHEAD, 32, 2 * CEMB);
      tdm_load_2d(offV[buf ^ 1], vheadT + kv0 + 32, 32, DHEAD, TKV);
    }

    // ---- scores S[16 x 32] from LDS K tile ----
    v8f s0 = {}, s1 = {};
    {
      const __bf16* kr0 = &Kst[buf][ml][kg * 16];
      const __bf16* kr1 = &Kst[buf][16 + ml][kg * 16];
      v16bf bk00 = *(const v16bf*)(kr0);        // kv cols +0..15, Dh 0..31
      v16bf bk01 = *(const v16bf*)(kr0 + 32);   // kv cols +0..15, Dh 32..63
      v16bf bk10 = *(const v16bf*)(kr1);
      v16bf bk11 = *(const v16bf*)(kr1 + 32);
      s0 = WMMA_BF16(qa0, bk00, s0);
      s0 = WMMA_BF16(qa1, bk01, s0);
      s1 = WMMA_BF16(qa0, bk10, s1);
      s1 = WMMA_BF16(qa1, bk11, s1);
    }

    // ---- online softmax (row stats per-lane for rows r + 8*kg) ----
    float alpha[8];
#pragma unroll
    for (int r = 0; r < 8; ++r) {
      float v = reduce16_max(fmaxf(s0[r], s1[r]) * 8.0f);
      float mnew = fmaxf(mrow[r], v);
      alpha[r] = __expf(mrow[r] - mnew);
      mrow[r] = mnew;
    }
#pragma unroll
    for (int r = 0; r < 8; ++r) {
      float p0 = __expf(s0[r] * 8.0f - mrow[r]);
      float p1 = __expf(s1[r] * 8.0f - mrow[r]);
      const int m = r + 8 * kg;
      Pst[wave][m][ml] = (__bf16)p0;
      Pst[wave][m][16 + ml] = (__bf16)p1;
      lrow[r] = lrow[r] * alpha[r] + reduce16_sum(p0 + p1);
    }
#pragma unroll
    for (int r = 0; r < 8; ++r) {  // rescale running output
      o0[r] *= alpha[r]; o1[r] *= alpha[r];
      o2[r] *= alpha[r]; o3[r] *= alpha[r];
    }

    // ---- reshape P into A-fragment layout via LDS (wave-private) ----
    v16bf pa;
#pragma unroll
    for (int e = 0; e < 16; ++e) {
      int k = kg * 16 + e;
      pa[e] = Pst[wave][ml][k];
    }

    // ---- O += P @ V from LDS V^T tile ----
    v16bf bv0 = *(const v16bf*)&Vst[buf][0 * 16 + ml][kg * 16];
    v16bf bv1 = *(const v16bf*)&Vst[buf][1 * 16 + ml][kg * 16];
    v16bf bv2 = *(const v16bf*)&Vst[buf][2 * 16 + ml][kg * 16];
    v16bf bv3 = *(const v16bf*)&Vst[buf][3 * 16 + ml][kg * 16];
    o0 = WMMA_BF16(pa, bv0, o0);
    o1 = WMMA_BF16(pa, bv1, o1);
    o2 = WMMA_BF16(pa, bv2, o2);
    o3 = WMMA_BF16(pa, bv3, o3);

    __syncthreads();  // reads of KV tile(kv0) done before overwrite
  }

  // ---- epilogue: y = O / l ----
#pragma unroll
  for (int r = 0; r < 8; ++r) {
    const float inv = 1.0f / lrow[r];
    const size_t row = (size_t)(b * TQ + q0 + r + 8 * kg) * CEMB + h * DHEAD;
    yb[row + 0 * 16 + ml] = (__bf16)(o0[r] * inv);
    yb[row + 1 * 16 + ml] = (__bf16)(o1[r] * inv);
    yb[row + 2 * 16 + ml] = (__bf16)(o2[r] * inv);
    yb[row + 3 * 16 + ml] = (__bf16)(o3[r] * inv);
  }
}

// ---------------------------------------------------------------------------
// Host-side orchestration
// ---------------------------------------------------------------------------
extern "C" void kernel_launch(void* const* d_in, const int* in_sizes, int n_in,
                              void* d_out, int out_size, void* d_ws, size_t ws_size,
                              hipStream_t stream) {
  (void)in_sizes; (void)n_in; (void)out_size; (void)ws_size;

  const float* x_q    = (const float*)d_in[0];
  const float* x_kv   = (const float*)d_in[1];
  const float* W_q    = (const float*)d_in[2];
  const float* W_kv   = (const float*)d_in[3];
  const float* W_proj = (const float*)d_in[4];
  float* out = (float*)d_out;

  const size_t MB = 1024ull * 1024ull;
  char* ws = (char*)d_ws;
  __bf16* xq_bf  = (__bf16*)(ws + 0 * MB);   // 8 MB  [B*TQ, C]
  __bf16* xkv_bf = (__bf16*)(ws + 8 * MB);   // 8 MB  [B*TKV, C]
  __bf16* WqT    = (__bf16*)(ws + 16 * MB);  // 2 MB  [C, C]
  __bf16* WkvT   = (__bf16*)(ws + 18 * MB);  // 4 MB  [2C, C]
  __bf16* WpT    = (__bf16*)(ws + 22 * MB);  // 2 MB  [C, C]
  __bf16* q_bf   = (__bf16*)(ws + 24 * MB);  // 8 MB  [B*TQ, C]
  __bf16* kv_bf  = (__bf16*)(ws + 32 * MB);  // 16 MB [B*TKV, 2C]
  __bf16* vT_bf  = (__bf16*)(ws + 48 * MB);  // 8 MB  [B,H,Dh,TKV]
  __bf16* y_bf   = (__bf16*)(ws + 56 * MB);  // 8 MB  [B*TQ, C]

  const int nX = BATCH * TQ * CEMB;  // 4,194,304

  // 1. converts
  cvt_f32_bf16<<<(nX + 255) / 256, 256, 0, stream>>>(x_q, xq_bf, nX);
  cvt_f32_bf16<<<(nX + 255) / 256, 256, 0, stream>>>(x_kv, xkv_bf, nX);
  transpose_cvt<<<(CEMB * CEMB + 255) / 256, 256, 0, stream>>>(W_q, WqT, CEMB, CEMB);
  transpose_cvt<<<(CEMB * 2 * CEMB + 255) / 256, 256, 0, stream>>>(W_kv, WkvT, CEMB, 2 * CEMB);
  transpose_cvt<<<(CEMB * CEMB + 255) / 256, 256, 0, stream>>>(W_proj, WpT, CEMB, CEMB);

  // 2. q = x_q @ W_q   (M=4096, N=1024, K=1024) -> bf16
  {
    dim3 grid(CEMB / 64, (BATCH * TQ) / 128);
    gemm_bf16_wmma<false><<<grid, 256, 0, stream>>>(xq_bf, WqT, q_bf,
                                                    BATCH * TQ, CEMB, CEMB);
  }
  // 3. kv = x_kv @ W_kv (M=4096, N=2048, K=1024) -> bf16
  {
    dim3 grid((2 * CEMB) / 64, (BATCH * TKV) / 128);
    gemm_bf16_wmma<false><<<grid, 256, 0, stream>>>(xkv_bf, WkvT, kv_bf,
                                                    BATCH * TKV, 2 * CEMB, CEMB);
  }
  // 4. V^T
  {
    int n = BATCH * NHEAD * DHEAD * TKV;
    make_vT<<<(n + 255) / 256, 256, 0, stream>>>(kv_bf, vT_bf);
  }
  // 5. attention
  {
    dim3 grid(TQ / 128, NHEAD, BATCH);
    attn_wmma<<<grid, 256, 0, stream>>>(q_bf, kv_bf, vT_bf, y_bf);
  }
  // 6. out = y @ W_proj (M=4096, N=1024, K=1024) -> f32 into d_out
  {
    dim3 grid(CEMB / 64, (BATCH * TQ) / 128);
    gemm_bf16_wmma<true><<<grid, 256, 0, stream>>>(y_bf, WpT, out,
                                                   BATCH * TQ, CEMB, CEMB);
  }
}
